// DownSampling_17987323036116
// MI455X (gfx1250) — compile-verified, matrix operand order checked by reading, not run on hardware
//
#include <hip/hip_runtime.h>
#include <hip/hip_bf16.h>

#define B_DIM 32768
#define C_DIM 512
#define CH    1024                 // rows staged per TDM chunk
#define NCH   (B_DIM / CH)

typedef __attribute__((ext_vector_type(2)))  float        v2f;
typedef __attribute__((ext_vector_type(8)))  float        v8f;
typedef __attribute__((ext_vector_type(16))) _Float16     v16h;
typedef __attribute__((ext_vector_type(4)))  unsigned int u32x4;
typedef __attribute__((ext_vector_type(4)))  int          i32x4;
typedef __attribute__((ext_vector_type(8)))  int          i32x8;

#define TDM_OK (__has_builtin(__builtin_amdgcn_tensor_load_to_lds) && \
                __has_builtin(__builtin_amdgcn_s_wait_tensorcnt))

// ---------------------------------------------------------------------------
// WMMA block reduction (verified round 1: lowers to 4x v_wmma_f32_16x16x4_f32).
// ---------------------------------------------------------------------------
__device__ __forceinline__ float wave0_wmma_sum256(const float* red, int lane) {
#if __has_builtin(__builtin_amdgcn_wmma_f32_16x16x4_f32)
  v8f d = {};
  v2f a; a[0] = 1.0f; a[1] = 1.0f;
#pragma unroll
  for (int k = 0; k < 4; ++k) {
    v2f b; b[0] = red[64 * k + lane]; b[1] = red[64 * k + 32 + lane];
    d = __builtin_amdgcn_wmma_f32_16x16x4_f32(false, a, false, b, (short)0, d,
                                              false, false);
  }
  return d[0];
#elif __has_builtin(__builtin_amdgcn_wmma_f32_16x16x32_f16)
  v16h a, b;
#pragma unroll
  for (int k = 0; k < 16; ++k) a[k] = (_Float16)1.0f;
#pragma unroll
  for (int k = 0; k < 8; ++k) b[k] = (_Float16)red[32 * k + lane];
#pragma unroll
  for (int k = 8; k < 16; ++k) b[k] = (_Float16)0.0f;
  v8f d = {};
  d = __builtin_amdgcn_wmma_f32_16x16x32_f16(false, a, false, b, (short)0, d,
                                             false, false);
  return d[0];
#else
  float s = 0.0f;
  for (int k = 0; k < 8; ++k) s += red[32 * k + lane];
  return 2.0f * s;  // pre-compensate shared *0.5
#endif
}

__device__ __forceinline__ float block_sum_256(float v, float* red) {
  red[threadIdx.x] = v;
  __syncthreads();
  if (threadIdx.x < 32) {  // wave 0 only: EXEC all-ones within the wave
    const int lane = (int)threadIdx.x;
    float s = wave0_wmma_sum256(red, lane);
#pragma unroll
    for (int off = 16; off > 0; off >>= 1) s += __shfl_xor(s, off, 32);
    s *= 0.5f;
    if (lane == 0) red[0] = s;
  }
  __syncthreads();
  const float r = red[0];
  __syncthreads();
  return r;
}

__device__ __forceinline__ float bce_logits(float x, float tf) {
  return fmaxf(x, 0.0f) - x * tf + log1pf(expf(-fabsf(x)));  // >= 0 always
}

#if TDM_OK
// ---------------------------------------------------------------------------
// TDM column gather: 2D tile, tile_dim0 = 1 element (4B), tile_dim1 = rows,
// row stride = C_DIM elements -> lands contiguous in LDS. Async (TENSORcnt).
// ---------------------------------------------------------------------------
__device__ __forceinline__ void tdm_load_column(const void* gptr, void* lptr,
                                                int rows) {
  const unsigned           lds = (unsigned)(uintptr_t)lptr;  // low 32 = LDS off
  const unsigned long long ga  = (unsigned long long)(uintptr_t)gptr;
  u32x4 g0;
  g0[0] = 1u;                                        // count=1, user descriptor
  g0[1] = lds;                                       // lds_addr
  g0[2] = (unsigned)(ga & 0xFFFFFFFFu);              // global_addr[31:0]
  g0[3] = (unsigned)((ga >> 32) & 0x1FFFFFFu) | (2u << 30);  // [56:32] | type=2
  i32x8 g1;
  g1[0] = (int)(2u << 16);                           // data_size = 4B
  g1[1] = (int)(1u << 16);                           // tensor_dim0 = 1
  g1[2] = (int)(((unsigned)rows & 0xFFFFu) << 16);   // tensor_dim1.lo = rows
  g1[3] = (int)(1u << 16);                           // dim1.hi=0 | tile_dim0=1
  g1[4] = (int)((unsigned)rows & 0xFFFFu);           // tile_dim1 | tile_dim2=0
  g1[5] = (int)C_DIM;                                // tensor_dim0_stride = 512
  g1[6] = 0;                                         // stride0.hi | stride1.lo
  g1[7] = 0;                                         // stride1.hi
  const i32x4 z4 = {0, 0, 0, 0};
#if defined(__clang_major__) && (__clang_major__ >= 23)
  const i32x8 z8 = {0, 0, 0, 0, 0, 0, 0, 0};
  __builtin_amdgcn_tensor_load_to_lds(g0, g1, z4, z4, z8, 0);
#else
  __builtin_amdgcn_tensor_load_to_lds(g0, g1, z4, z4, 0);
#endif
}

// One workgroup per class column. TDM double-buffers the strided column into
// LDS once; keys (= bce bits | label<<31, valid since bce >= 0) stay LDS-
// resident for the 3-pass radix select. Global memory is read exactly once.
__global__ __launch_bounds__(256) void downsample_column_kernel(
    const float* __restrict__ pred, const int* __restrict__ tgt,
    float* __restrict__ colsum) {
  __shared__ union {
    struct { float p[2][CH]; int t[2][CH]; } st;          // 16 KB staging
    struct { unsigned cnt[2048]; float sum[2048]; } h;    // 16 KB histograms
  } u;
  __shared__ unsigned keys[B_DIM];                        // 128 KB
  __shared__ float    red[256];
  __shared__ unsigned s_prefix;
  __shared__ int      s_mrem;
  __shared__ float    s_part;

  const int c   = (int)blockIdx.x;
  const int tid = (int)threadIdx.x;
  const float* colp = pred + c;
  const int*   colt = tgt + c;

  // ---- Phase 0: stage column once, build keys, count positives -------------
  if (tid < 32) {  // wave 0 issues all TDM traffic
    tdm_load_column(colp, &u.st.p[0][0], CH);
    tdm_load_column(colt, &u.st.t[0][0], CH);
  }
  int cnt = 0;
  for (int ch = 0; ch < NCH; ++ch) {
    const int b = ch & 1;
    if (tid < 32) {
      if (ch + 1 < NCH) {
        tdm_load_column(colp + (size_t)(ch + 1) * CH * C_DIM, &u.st.p[b ^ 1][0], CH);
        tdm_load_column(colt + (size_t)(ch + 1) * CH * C_DIM, &u.st.t[b ^ 1][0], CH);
        __builtin_amdgcn_s_wait_tensorcnt((short)2);  // current chunk complete
      } else {
        __builtin_amdgcn_s_wait_tensorcnt((short)0);
      }
    }
    __syncthreads();
    for (int i = tid; i < CH; i += 256) {
      const float x  = u.st.p[b][i];
      const int   ti = u.st.t[b][i];
      const float bce = bce_logits(x, (float)ti);
      keys[ch * CH + i] = __float_as_uint(bce) | ((unsigned)ti << 31);
      cnt += ti;
    }
    __syncthreads();
  }
  const float pos_f   = block_sum_256((float)cnt, red);
  const int   pos_sum = (int)(pos_f + 0.5f);
  const int   pos_gt  = (2 * pos_sum >= B_DIM) ? 1 : 0;
  const int   m       = pos_gt ? (B_DIM - pos_sum) : pos_sum;

  if (m == 0) { if (tid == 0) colsum[c] = 0.0f; return; }
  if (tid == 0) { s_prefix = 0u; s_mrem = m; s_part = 0.0f; }

  float minority_sum = 0.0f;

  // ---- 3-pass radix select entirely in LDS ---------------------------------
  for (int pass = 0; pass < 3; ++pass) {
    for (int i = tid; i < 2048; i += 256) { u.h.cnt[i] = 0u; u.h.sum[i] = 0.0f; }
    __syncthreads();
    const unsigned prefix = s_prefix;
    const unsigned pmask  = (pass == 0) ? 0u
                          : (pass == 1) ? 0xFFE00000u : 0xFFFFFC00u;
    const int      shift  = (pass == 0) ? 21 : (pass == 1) ? 10 : 0;
    const unsigned dmask  = (pass == 2) ? 0x3FFu : 0x7FFu;

    for (int r = tid; r < B_DIM; r += 256) {
      const unsigned w        = keys[r];
      const int      ti       = (int)(w >> 31);
      const unsigned bcebits  = w & 0x7FFFFFFFu;
      const bool     maj      = (ti == pos_gt);
      const unsigned key      = maj ? bcebits : 0u;
      if (pass == 0 && !maj) minority_sum += __uint_as_float(bcebits);
      if ((key & pmask) == prefix) {
        const unsigned d = (key >> shift) & dmask;
        atomicAdd(&u.h.cnt[d], 1u);
        atomicAdd(&u.h.sum[d], __uint_as_float(key));  // 0 for minority
      }
    }
    __syncthreads();
    if (tid == 0) {
      int      mrem = s_mrem;
      float    part = s_part;
      int      d    = (pass == 2) ? 1023 : 2047;
      unsigned cum  = 0;
      while (d > 0 && cum + u.h.cnt[d] < (unsigned)mrem) {
        cum += u.h.cnt[d];
        part += u.h.sum[d];
        --d;
      }
      s_prefix = s_prefix | ((unsigned)d << shift);
      s_mrem   = mrem - (int)cum;
      s_part   = part;
    }
    __syncthreads();
  }

  const float tau    = __uint_as_float(s_prefix);
  const float tie    = (float)s_mrem;
  const float sel_hi = s_part;
  const float msum   = block_sum_256(minority_sum, red);
  if (tid == 0) colsum[c] = msum + sel_hi + tau * tie;
}

#else  // -------- fallback: round-1 kernel (direct global reads + prefetch) --

__global__ __launch_bounds__(256) void downsample_column_kernel(
    const float* __restrict__ pred, const int* __restrict__ tgt,
    float* __restrict__ colsum) {
  __shared__ unsigned hcnt[2048];
  __shared__ float    hsum[2048];
  __shared__ float    red[256];
  __shared__ unsigned s_prefix;
  __shared__ int      s_mrem;
  __shared__ float    s_part;

  const int c   = (int)blockIdx.x;
  const int tid = (int)threadIdx.x;

  int cnt = 0;
  for (int r = tid; r < B_DIM; r += 256) cnt += tgt[r * C_DIM + c];
  const float pos_f   = block_sum_256((float)cnt, red);
  const int   pos_sum = (int)(pos_f + 0.5f);
  const int   pos_gt  = (2 * pos_sum >= B_DIM) ? 1 : 0;
  const int   m       = pos_gt ? (B_DIM - pos_sum) : pos_sum;

  if (m == 0) { if (tid == 0) colsum[c] = 0.0f; return; }
  if (tid == 0) { s_prefix = 0u; s_mrem = m; s_part = 0.0f; }

  float minority_sum = 0.0f;
  for (int pass = 0; pass < 3; ++pass) {
    for (int i = tid; i < 2048; i += 256) { hcnt[i] = 0u; hsum[i] = 0.0f; }
    __syncthreads();
    const unsigned prefix = s_prefix;
    const unsigned pmask  = (pass == 0) ? 0u
                          : (pass == 1) ? 0xFFE00000u : 0xFFFFFC00u;
    const int      shift  = (pass == 0) ? 21 : (pass == 1) ? 10 : 0;
    const unsigned dmask  = (pass == 2) ? 0x3FFu : 0x7FFu;

    for (int r = tid; r < B_DIM; r += 256) {
      const int idx = r * C_DIM + c;
      if (r + 256 < B_DIM) {
        __builtin_prefetch(pred + idx + 256 * C_DIM, 0, 1);
        __builtin_prefetch(tgt + idx + 256 * C_DIM, 0, 1);
      }
      const float    x   = pred[idx];
      const int      ti  = tgt[idx];
      const float    bce = bce_logits(x, (float)ti);
      const bool     maj = (ti == pos_gt);
      const unsigned key = maj ? __float_as_uint(bce) : 0u;
      if (pass == 0 && !maj) minority_sum += bce;
      if ((key & pmask) == prefix) {
        const unsigned d = (key >> shift) & dmask;
        atomicAdd(&hcnt[d], 1u);
        atomicAdd(&hsum[d], __uint_as_float(key));
      }
    }
    __syncthreads();
    if (tid == 0) {
      int      mrem = s_mrem;
      float    part = s_part;
      int      d    = (pass == 2) ? 1023 : 2047;
      unsigned cum  = 0;
      while (d > 0 && cum + hcnt[d] < (unsigned)mrem) {
        cum += hcnt[d];
        part += hsum[d];
        --d;
      }
      s_prefix = s_prefix | ((unsigned)d << shift);
      s_mrem   = mrem - (int)cum;
      s_part   = part;
    }
    __syncthreads();
  }

  const float tau    = __uint_as_float(s_prefix);
  const float tie    = (float)s_mrem;
  const float sel_hi = s_part;
  const float msum   = block_sum_256(minority_sum, red);
  if (tid == 0) colsum[c] = msum + sel_hi + tau * tie;
}
#endif  // TDM_OK

__global__ __launch_bounds__(256) void finalize_kernel(
    const float* __restrict__ colsum, float* __restrict__ out) {
  __shared__ float red[256];
  const int t = (int)threadIdx.x;
  const float v = colsum[t] + colsum[t + 256];
  const float total = block_sum_256(v, red);
  if (t == 0) out[0] = total * (1.0f / ((float)B_DIM * (float)C_DIM));
}

extern "C" void kernel_launch(void* const* d_in, const int* in_sizes, int n_in,
                              void* d_out, int out_size, void* d_ws,
                              size_t ws_size, hipStream_t stream) {
  (void)in_sizes; (void)n_in; (void)out_size; (void)ws_size;
  const float* pred = (const float*)d_in[0];
  const int*   tgt  = (const int*)d_in[1];
  float* colsum = (float*)d_ws;  // 512 floats, fully rewritten every launch

  downsample_column_kernel<<<C_DIM, 256, 0, stream>>>(pred, tgt, colsum);
  finalize_kernel<<<1, 256, 0, stream>>>(colsum, (float*)d_out);
}
